// CasualAttention_42966852829217
// MI455X (gfx1250) — compile-verified, matrix operand order checked by reading, not run on hardware
//
#include <hip/hip_runtime.h>
#include <hip/hip_bf16.h>

// ---------------------------------------------------------------------------
// CDNA5 (gfx1250) fused causal attention:
//   K1: fused QKV projection GEMM (fp32 -> bf16, WMMA f32_16x16x32_bf16)
//   K2: flash attention, 320KB-LDS-resident tiles, online softmax,
//       async global->LDS staging (GLOBAL_LOAD_ASYNC_TO_LDS_B128 / ASYNCcnt).
// Workspace layout (48 MB, rewritten every call):
//   Qb : bf16 [4][2048][1024]
//   Kb : bf16 [4][2048][1024]
//   Vt : bf16 [4][1024][2048]   (V pre-transposed for contiguous B-fragments)
// ---------------------------------------------------------------------------

typedef __attribute__((ext_vector_type(16))) __bf16 v16bf;
typedef __attribute__((ext_vector_type(8)))  __bf16 v8bf;
typedef __attribute__((ext_vector_type(8)))  float  v8f;
typedef int v4i __attribute__((__vector_size__(16)));

__device__ __forceinline__ __bf16 f2bf(float x) {
  unsigned u = __builtin_bit_cast(unsigned, x);
  unsigned r = u + 0x7FFFu + ((u >> 16) & 1u);   // round-to-nearest-even
  unsigned short h = (unsigned short)(r >> 16);
  return __builtin_bit_cast(__bf16, h);
}

// Pack two fp32 -> one dword of 2x bf16 (v_cvt_pk_bf16_f32 when available).
__device__ __forceinline__ unsigned pack2bf(float a, float b) {
#if __has_builtin(__builtin_amdgcn_cvt_pk_bf16_f32)
  typedef __attribute__((ext_vector_type(2))) __bf16 v2bf;
  v2bf r = __builtin_amdgcn_cvt_pk_bf16_f32(a, b);
  return __builtin_bit_cast(unsigned, r);
#else
  unsigned ua = __builtin_bit_cast(unsigned, a);
  unsigned ub = __builtin_bit_cast(unsigned, b);
  ua = (ua + 0x7FFFu + ((ua >> 16) & 1u)) >> 16;
  ub = (ub + 0x7FFFu + ((ub >> 16) & 1u)) & 0xFFFF0000u;
  return ua | ub;
#endif
}

// Async global->LDS 16-byte copy (GLOBAL_LOAD_ASYNC_TO_LDS_B128, ASYNCcnt),
// with a VGPR-staged fallback if the builtin is not declared.
#if __has_builtin(__builtin_amdgcn_global_load_async_to_lds_b128)
#define ASYNC_LDS 1
#else
#define ASYNC_LDS 0
#endif

__device__ __forceinline__ void cp16(__bf16* dst, const __bf16* src) {
#if ASYNC_LDS
  __builtin_amdgcn_global_load_async_to_lds_b128(
      (__attribute__((address_space(1))) v4i*)(void*)src,
      (__attribute__((address_space(3))) v4i*)(void*)dst, 0, 0);
#else
  *(v8bf*)dst = *(const v8bf*)src;
#endif
}

__device__ __forceinline__ void cp_wait() {
#if ASYNC_LDS
#if __has_builtin(__builtin_amdgcn_s_wait_asynccnt)
  __builtin_amdgcn_s_wait_asynccnt(0);
#else
  asm volatile("s_wait_asynccnt 0x0" ::: "memory");
#endif
#endif
}

__device__ __forceinline__ v8f wmma_bf16(v16bf a, v16bf b, v8f c) {
  // (neg_a, A, neg_b, B, c_mod, C, reuse_a, reuse_b)
  return __builtin_amdgcn_wmma_f32_16x16x32_bf16(false, a, false, b, (short)0, c,
                                                 false, false);
}

// A/B fragment loader from an LDS tile whose "rows" are contiguous in K.
// ISA 7.12.2 16-bit A layout: lanes 0-15 hold row r, K = {kb..kb+7, kb+16..kb+23};
// lanes 16-31 hold row r, K = {kb+8..kb+15, kb+24..kb+31}.  B fragments are the
// same pattern applied to B^T (lane holds one column of B).
__device__ __forceinline__ v16bf load_frag(const __bf16* base, int stride,
                                           int row0, int kb, int lane) {
  const int r   = lane & 15;
  const int sel = (lane >> 4) << 3;              // 0 or 8
  const __bf16* p = base + (row0 + r) * stride + kb + sel;
  union { v16bf f; v8bf h[2]; } u;
  u.h[0] = *(const v8bf*)(p);                    // ds_load_b128
  u.h[1] = *(const v8bf*)(p + 16);               // ds_load_b128
  return u.f;
}

// ---------------------------------------------------------------------------
// Kernel 1: fused QKV projection.  out[z] = x @ W[z]  (M=8192, N=1024, K=1024)
// grid = (N/64, M/128, 3), block = 256 (8 waves).  Wave (wm,wn) owns 32x32.
// BLOCK_K = 64 -> 8 WMMAs per staging round.
// ---------------------------------------------------------------------------
__global__ __launch_bounds__(256) void qkv_proj_kernel(
    const float* __restrict__ x,
    const float* __restrict__ Wq, const float* __restrict__ Wk,
    const float* __restrict__ Wv,
    __bf16* __restrict__ Qb, __bf16* __restrict__ Kb, __bf16* __restrict__ Vt) {
  __shared__ __bf16 sA[128 * 64];   // 16 KB  x tile   [m][k]
  __shared__ __bf16 sBt[64 * 64];   //  8 KB  W tile transposed: [n][k]

  const int tid  = threadIdx.x;
  const int lane = tid & 31;
  const int w    = tid >> 5;
  const int wm   = w >> 1;          // 0..3
  const int wn   = w & 1;           // 0..1
  const int n0   = blockIdx.x * 64;
  const int m0   = blockIdx.y * 128;
  const int z    = blockIdx.z;
  const float* W = (z == 0) ? Wq : (z == 1) ? Wk : Wv;

  v8f acc[2][2];
#pragma unroll
  for (int i = 0; i < 2; i++)
#pragma unroll
    for (int j = 0; j < 2; j++) acc[i][j] = v8f{};

  for (int kb0 = 0; kb0 < 1024; kb0 += 64) {
    // Stage A: 128x64 fp32 -> bf16 (2048 float4, 8 per thread, packed b64 DS)
#pragma unroll
    for (int i = 0; i < 8; i++) {
      int idx = tid + i * 256;                 // float4 index
      int r   = idx >> 4;                      // 16 float4 per row of 64
      int c4  = idx & 15;
      float4 f = *(const float4*)(x + (size_t)(m0 + r) * 1024 + kb0 + c4 * 4);
      uint2 p;
      p.x = pack2bf(f.x, f.y);
      p.y = pack2bf(f.z, f.w);
      *(uint2*)(sA + r * 64 + c4 * 4) = p;
    }
    // Stage B transposed: W[kb0+k][n0+n] -> sBt[n][k] (1024 float4, 4/thread)
#pragma unroll
    for (int i = 0; i < 4; i++) {
      int idx = tid + i * 256;
      int k   = idx >> 4;                      // 16 float4 per row of 64
      int c4  = idx & 15;
      float4 f = *(const float4*)(W + (size_t)(kb0 + k) * 1024 + n0 + c4 * 4);
      int n = c4 * 4;
      sBt[(n + 0) * 64 + k] = f2bf(f.x);
      sBt[(n + 1) * 64 + k] = f2bf(f.y);
      sBt[(n + 2) * 64 + k] = f2bf(f.z);
      sBt[(n + 3) * 64 + k] = f2bf(f.w);
    }
    __syncthreads();

#pragma unroll
    for (int ks = 0; ks < 2; ks++) {
      int kb = ks * 32;
      v16bf a0 = load_frag(sA, 64, wm * 32 + 0,  kb, lane);
      v16bf a1 = load_frag(sA, 64, wm * 32 + 16, kb, lane);
      v16bf b0 = load_frag(sBt, 64, wn * 32 + 0,  kb, lane);
      v16bf b1 = load_frag(sBt, 64, wn * 32 + 16, kb, lane);
      acc[0][0] = wmma_bf16(a0, b0, acc[0][0]);
      acc[0][1] = wmma_bf16(a0, b1, acc[0][1]);
      acc[1][0] = wmma_bf16(a1, b0, acc[1][0]);
      acc[1][1] = wmma_bf16(a1, b1, acc[1][1]);
    }
    __syncthreads();
  }

  // Epilogue.  C layout: lane<16 -> M=v, lane>=16 -> M=v+8; N = lane&15.
  const int r   = lane & 15;
  const int sel = lane >> 4;
  if (z == 2) {
    // Vt[b][d][s]: the 8 accumulator rows are contiguous in s -> b128 stores.
#pragma unroll
    for (int mt = 0; mt < 2; mt++)
#pragma unroll
      for (int nt = 0; nt < 2; nt++) {
        uint4 pk;
        pk.x = pack2bf(acc[mt][nt][0], acc[mt][nt][1]);
        pk.y = pack2bf(acc[mt][nt][2], acc[mt][nt][3]);
        pk.z = pack2bf(acc[mt][nt][4], acc[mt][nt][5]);
        pk.w = pack2bf(acc[mt][nt][6], acc[mt][nt][7]);
        int rowb = m0 + wm * 32 + mt * 16 + sel * 8;   // row of v=0
        int col  = n0 + wn * 32 + nt * 16 + r;
        int bb = rowb >> 11, s0 = rowb & 2047;
        *(uint4*)(Vt + ((size_t)bb * 1024 + col) * 2048 + s0) = pk;
      }
  } else {
    __bf16* __restrict__ dst = (z == 0) ? Qb : Kb;
#pragma unroll
    for (int mt = 0; mt < 2; mt++)
#pragma unroll
      for (int nt = 0; nt < 2; nt++)
#pragma unroll
        for (int v = 0; v < 8; v++) {
          int row = m0 + wm * 32 + mt * 16 + v + sel * 8;
          int col = n0 + wn * 32 + nt * 16 + r;
          dst[(size_t)row * 1024 + col] = f2bf(acc[mt][nt][v]);
        }
  }
}

// ---------------------------------------------------------------------------
// Kernel 2: flash attention.  grid = (2048/64, 4), block = 512 (16 waves).
// Q tile (64x1024 bf16) resident in LDS; iterate 32-key tiles up to diagonal.
// Wave w owns O columns [w*64, w*64+64) as 4x4 fp32 WMMA accumulators.
// ---------------------------------------------------------------------------
__global__ __launch_bounds__(512) void flash_attn_kernel(
    const __bf16* __restrict__ Qb, const __bf16* __restrict__ Kb,
    const __bf16* __restrict__ Vt, float* __restrict__ out) {
  __shared__ __bf16 sQ[64 * 1024];    // 128 KB  [q][d]
  __shared__ __bf16 sK[32 * 1024];    //  64 KB  [k][d]
  __shared__ __bf16 sV[1024 * 32];    //  64 KB  [d][k]  (transposed tile)
  __shared__ float  sS0[64 * 32];     //   8 KB  S partial (k-half 0)
  __shared__ float  sS1[64 * 32];     //   8 KB  S partial (k-half 1)
  __shared__ __bf16 sP[64 * 32];      //   4 KB  softmaxed probs
  __shared__ float  sScale[64];
  __shared__ float  sL[64];

  const int tid   = threadIdx.x;
  const int lane  = tid & 31;
  const int w     = tid >> 5;                  // 0..15
  const int qt    = blockIdx.x;
  const int b     = blockIdx.y;
  const int qbase = qt * 64;
  const __bf16* Qg = Qb + ((size_t)b * 2048 + qbase) * 1024;
  const __bf16* Kg = Kb + (size_t)b * 2048 * 1024;
  const __bf16* Vg = Vt + (size_t)b * 1024 * 2048;

  // Load resident Q tile (8192 x 16B vectors, 16 per thread) - async to LDS.
#pragma unroll
  for (int i = 0; i < 16; i++) {
    int idx = tid + i * 512;
    cp16(sQ + idx * 8, Qg + idx * 8);
  }

  // S-phase role: 8 score tiles x 2 k-halves across 16 waves.
  const int mt = (w & 7) >> 1;                 // 0..3 : query 16-row tile
  const int nt = w & 1;                        // 0..1 : key 16-col tile
  const int kh = w >> 3;                       // 0..1 : which 512-d half

  v8f o[4][4];
#pragma unroll
  for (int i = 0; i < 4; i++)
#pragma unroll
    for (int j = 0; j < 4; j++) o[i][j] = v8f{};

  float mrun = -__builtin_inff();              // row-owner state (tid < 64)
  float lrun = 0.f;

  const int nkt = 2 * qt + 2;                  // causal: only tiles <= diagonal
  for (int j = 0; j < nkt; j++) {
    const int k0 = j * 32;
    cp_wait();
    __syncthreads();                           // prev PV done with sK/sV/sP

    // Stage K tile [32][1024] (4096 vec8, 8 per thread) - async to LDS.
#pragma unroll
    for (int i = 0; i < 8; i++) {
      int idx = tid + i * 512;
      int key = idx >> 7, c = idx & 127;
      cp16(sK + key * 1024 + c * 8, Kg + (size_t)(k0 + key) * 1024 + c * 8);
    }
    // Stage V^T tile [1024][32] (4096 vec8, 8 per thread) - async to LDS.
#pragma unroll
    for (int i = 0; i < 8; i++) {
      int idx = tid + i * 512;
      int d = idx >> 2, c = idx & 3;
      cp16(sV + d * 32 + c * 8, Vg + (size_t)d * 2048 + k0 + c * 8);
    }
    cp_wait();
    __syncthreads();

    // Prefetch next key tile into cache while this one is computed.
    if (j + 1 < nkt) {
      const int k0n = k0 + 32;
      const int row = tid >> 4, c = tid & 15;  // 32 rows x 16 x 128B = 64KB
      __builtin_prefetch(Kg + (size_t)(k0n + row) * 1024 + c * 64, 0, 0);
      __builtin_prefetch(Vg + (size_t)(tid * 2) * 2048 + k0n, 0, 0);
      __builtin_prefetch(Vg + (size_t)(tid * 2 + 1) * 2048 + k0n, 0, 0);
    }

    // S = Q K^T over this wave's 512-d half (16 WMMA k-steps).
    v8f s = v8f{};
#pragma unroll
    for (int i = 0; i < 16; i++) {
      int kb = kh * 512 + i * 32;
      v16bf aq = load_frag(sQ, 1024, mt * 16, kb, lane);
      v16bf bk = load_frag(sK, 1024, nt * 16, kb, lane);
      s = wmma_bf16(aq, bk, s);
    }
    {
      float* Sd = kh ? sS1 : sS0;
      const int r = lane & 15, sel = lane >> 4;
#pragma unroll
      for (int v = 0; v < 8; v++)
        Sd[(mt * 16 + v + sel * 8) * 32 + nt * 16 + r] = s[v];
    }
    __syncthreads();

    // Online softmax: one thread per query row (exp2 domain).
    if (tid < 64) {
      const float cvt = 0.03125f * 1.44269504088896f;  // 1/sqrt(1024)*log2(e)
      const int qg = qbase + tid;
      float sv[32];
      float tm = -__builtin_inff();
#pragma unroll
      for (int kk = 0; kk < 32; kk++) {
        float xv = (sS0[tid * 32 + kk] + sS1[tid * 32 + kk]) * cvt;
        if (k0 + kk > qg) xv = -__builtin_inff();
        sv[kk] = xv;
        tm = fmaxf(tm, xv);
      }
      float mnew = fmaxf(mrun, tm);
      float sc   = exp2f(mrun - mnew);
      float sum  = 0.f;
#pragma unroll
      for (int kk = 0; kk < 32; kk += 2) {
        float p0 = exp2f(sv[kk]     - mnew);
        float p1 = exp2f(sv[kk + 1] - mnew);
        sum += p0 + p1;
        *(unsigned*)(sP + tid * 32 + kk) = pack2bf(p0, p1);
      }
      lrun = lrun * sc + sum;
      mrun = mnew;
      sScale[tid] = sc;
      sL[tid]     = lrun;
    }
    __syncthreads();

    // Rescale O accumulators, then O += P @ V (bf16 WMMA, one k-step of 32).
    {
      const int sel = lane >> 4;
#pragma unroll
      for (int m2 = 0; m2 < 4; m2++) {
        float f[8];
#pragma unroll
        for (int v = 0; v < 8; v++) f[v] = sScale[m2 * 16 + v + sel * 8];
#pragma unroll
        for (int n2 = 0; n2 < 4; n2++)
#pragma unroll
          for (int v = 0; v < 8; v++) o[m2][n2][v] *= f[v];
      }
      v16bf bv[4];
#pragma unroll
      for (int n2 = 0; n2 < 4; n2++)
        bv[n2] = load_frag(sV, 32, w * 64 + n2 * 16, 0, lane);
#pragma unroll
      for (int m2 = 0; m2 < 4; m2++) {
        v16bf ap = load_frag(sP, 32, m2 * 16, 0, lane);
#pragma unroll
        for (int n2 = 0; n2 < 4; n2++)
          o[m2][n2] = wmma_bf16(ap, bv[n2], o[m2][n2]);
      }
    }
  }

  // Epilogue: O / l, store fp32.
  {
    const int r = lane & 15, sel = lane >> 4;
#pragma unroll
    for (int m2 = 0; m2 < 4; m2++)
#pragma unroll
      for (int v = 0; v < 8; v++) {
        int row = m2 * 16 + v + sel * 8;
        float inv = 1.0f / sL[row];
#pragma unroll
        for (int n2 = 0; n2 < 4; n2++) {
          int col = w * 64 + n2 * 16 + r;
          out[((size_t)b * 2048 + qbase + row) * 1024 + col] = o[m2][n2][v] * inv;
        }
      }
  }
}

// ---------------------------------------------------------------------------
extern "C" void kernel_launch(void* const* d_in, const int* in_sizes, int n_in,
                              void* d_out, int out_size, void* d_ws,
                              size_t ws_size, hipStream_t stream) {
  (void)in_sizes; (void)n_in; (void)out_size; (void)ws_size;
  const float* x  = (const float*)d_in[0];
  const float* Wq = (const float*)d_in[1];
  const float* Wk = (const float*)d_in[2];
  const float* Wv = (const float*)d_in[3];
  float* out = (float*)d_out;

  const size_t BSD = (size_t)4 * 2048 * 1024;   // elements per QKV tensor
  __bf16* Qb = (__bf16*)d_ws;
  __bf16* Kb = Qb + BSD;
  __bf16* Vt = Kb + BSD;                        // 48 MB total workspace

  dim3 gProj(1024 / 64, 8192 / 128, 3);
  qkv_proj_kernel<<<gProj, 256, 0, stream>>>(x, Wq, Wk, Wv, Qb, Kb, Vt);

  dim3 gAttn(2048 / 64, 4, 1);
  flash_attn_kernel<<<gAttn, 512, 0, stream>>>(Qb, Kb, Vt, out);
}
